// BRNN_8151847927833
// MI455X (gfx1250) — 8-bit matrix operand order repaired and checked by reading, compiled, not run on hardware
//
#include <hip/hip_runtime.h>
#include <hip/hip_bf16.h>
#include <math.h>

typedef __attribute__((ext_vector_type(8))) int v8i_t;

#define NWG        32
#define NTHREADS   256
#define D_MODEL    1024
#define CARRY      896
#define READ_DIM   128
#define VOCAB      128
#define NUM_FF     6
#define BATCH      64
#define SEQ        512
#define LOGIT_SCALE 0.0625f

#define OFF_WT      0
#define OFF_EMBED   (6*1024*1024)
#define OFF_HEADT   (OFF_EMBED + 16384)
#define OFF_THR     (OFF_HEADT + 16384)
#define OFF_X0      (OFF_THR + 24576)
#define OFF_X1      (OFF_X0 + 65536)
#define OFF_READ    (OFF_X1 + 65536)
#define OFF_BAR     (OFF_READ + 8192)

__device__ __forceinline__ v8i_t load_fragA(const signed char* __restrict__ base,
                                            int row_stride, int row0, int k0) {
    const int lane = threadIdx.x & 31;
    const int half = lane >> 4;
    const int r = row0 + (lane & 15);
    const signed char* p = base + r * row_stride + k0 + half * 16;
    const int4 lo = *(const int4*)(p);
    const int4 hi = *(const int4*)(p + 32);
    v8i_t f;
    f[0]=lo.x; f[1]=lo.y; f[2]=lo.z; f[3]=lo.w;
    f[4]=hi.x; f[5]=hi.y; f[6]=hi.z; f[7]=hi.w;
    return f;
}
__device__ __forceinline__ v8i_t load_fragB(const signed char* __restrict__ base,
                                            int row_stride, int col0, int k0) {
    const int lane = threadIdx.x & 31;
    const int half = lane >> 4;
    const int n = col0 + (lane & 15);
    const signed char* p = base + n * row_stride + k0 + half * 16;
    const int4 lo = *(const int4*)(p);
    const int4 hi = *(const int4*)(p + 32);
    v8i_t f;
    f[0]=lo.x; f[1]=lo.y; f[2]=lo.z; f[3]=lo.w;
    f[4]=hi.x; f[5]=hi.y; f[6]=hi.z; f[7]=hi.w;
    return f;
}

__device__ __forceinline__ void grid_barrier(unsigned* cnt, unsigned* gen) {
    __threadfence();
    __syncthreads();
    if (threadIdx.x == 0) {
        unsigned g = __hip_atomic_load(gen, __ATOMIC_RELAXED, __HIP_MEMORY_SCOPE_AGENT);
        unsigned a = __hip_atomic_fetch_add(cnt, 1u, __ATOMIC_ACQ_REL, __HIP_MEMORY_SCOPE_AGENT);
        if (a == NWG - 1u) {
            __hip_atomic_store(cnt, 0u, __ATOMIC_RELAXED, __HIP_MEMORY_SCOPE_AGENT);
            __hip_atomic_store(gen, g + 1u, __ATOMIC_RELEASE, __HIP_MEMORY_SCOPE_AGENT);
        } else {
            while (__hip_atomic_load(gen, __ATOMIC_ACQUIRE, __HIP_MEMORY_SCOPE_AGENT) == g) {
                __builtin_amdgcn_s_sleep(2);
            }
        }
    }
    __syncthreads();
    __threadfence();
}

__global__ void init_barrier_k(unsigned* bar) { bar[0] = 0u; bar[1] = 0u; }

__global__ void prep_ff_k(const float* __restrict__ ff, signed char* __restrict__ wt) {
    int idx = blockIdx.x * NTHREADS + threadIdx.x;
    int l = idx >> 20;
    int rem = idx & 0xFFFFF;
    int n = rem >> 10;
    int k = rem & 1023;
    float v = ff[((long)l << 20) + ((long)k << 10) + n];
    wt[idx] = (v >= 0.0f) ? (signed char)1 : (signed char)-1;
}

__global__ void prep_small_k(const float* __restrict__ initial, const float* __restrict__ embed,
                             const float* __restrict__ head, const float* __restrict__ thresh,
                             signed char* __restrict__ embedS, signed char* __restrict__ headTS,
                             int* __restrict__ thrI, signed char* __restrict__ X0) {
    int idx = blockIdx.x * NTHREADS + threadIdx.x;
    if (idx < 16384) {
        embedS[idx] = (embed[idx] >= 0.0f) ? 1 : -1;
    } else if (idx < 32768) {
        int i = idx - 16384; int n = i >> 7, k = i & 127;
        headTS[i] = (head[k * 128 + n] >= 0.0f) ? 1 : -1;
    } else if (idx < 38912) {
        int i = idx - 32768;
        thrI[i] = (int)rintf(thresh[i]);
    } else if (idx < 104448) {
        int i = idx - 38912; int d = i & 1023;
        X0[i] = (initial[d] >= 0.0f) ? 1 : -1;
    }
}

__global__ __launch_bounds__(NTHREADS, 1)
void brnn_main_k(const int* __restrict__ tokens,
                 const signed char* __restrict__ Wt, const signed char* __restrict__ embedS,
                 const signed char* __restrict__ headTS, const int* __restrict__ thrI,
                 signed char* __restrict__ X0, signed char* __restrict__ X1,
                 signed char* __restrict__ READ,
                 unsigned* bar, float* __restrict__ out)
{
    __shared__ int   lds_logits[BATCH * VOCAB];
    __shared__ float lds_loss[BATCH];

    unsigned* cnt = bar;
    unsigned* gen = bar + 1;

    const int tid   = threadIdx.x;
    const int wg    = blockIdx.x;
    const int wave  = tid >> 5;
    const int lane  = tid & 31;
    const int half  = lane >> 4;
    const int l15   = lane & 15;
    const int mtile = wave & 3;
    const int ntile = wave >> 2;
    const int ncol  = wg * 32 + ntile * 16;

    float loss_acc = 0.0f;

    for (int t = 0; t < SEQ; ++t) {
        for (int l = 0; l < NUM_FF; ++l) {
            const signed char* Xin  = (l & 1) ? X1 : X0;
            signed char*       Xout = (l & 1) ? X0 : X1;
            const signed char* Wl   = Wt + ((long)l << 20);

            v8i_t acc = {};
            #pragma unroll 4
            for (int k = 0; k < D_MODEL; k += 64) {
                v8i_t a = load_fragA(Xin, D_MODEL, mtile * 16, k);
                v8i_t b = load_fragB(Wl,  D_MODEL, ncol,       k);
                acc = __builtin_amdgcn_wmma_i32_16x16x64_iu8(
                         true, a,  true, b, acc,
                         false,  false);
            }

            const int nc  = ncol + l15;
            const int thv = thrI[l * D_MODEL + nc];
            if (l < NUM_FF - 1) {
                #pragma unroll
                for (int j = 0; j < 8; ++j) {
                    int m = mtile * 16 + j + 8 * half;
                    Xout[m * D_MODEL + nc] = (acc[j] >= thv) ? 1 : -1;
                }
            } else {
                #pragma unroll
                for (int j = 0; j < 8; ++j) {
                    int m = mtile * 16 + j + 8 * half;
                    signed char v = (acc[j] >= thv) ? 1 : -1;
                    if (nc < CARRY) X0[m * D_MODEL + nc] = v;
                    else            READ[m * READ_DIM + (nc - CARRY)] = v;
                }
            }
            grid_barrier(cnt, gen);
        }

        if (wg == 0) {
            for (int nt = (wave >> 2); nt < 8; nt += 2) {
                v8i_t acc = {};
                v8i_t a0 = load_fragA(READ,  READ_DIM, mtile * 16, 0);
                v8i_t b0 = load_fragB(headTS, READ_DIM, nt * 16,   0);
                acc = __builtin_amdgcn_wmma_i32_16x16x64_iu8(true, a0, true, b0, acc, false, false);
                v8i_t a1 = load_fragA(READ,  READ_DIM, mtile * 16, 64);
                v8i_t b1 = load_fragB(headTS, READ_DIM, nt * 16,   64);
                acc = __builtin_amdgcn_wmma_i32_16x16x64_iu8(true, a1, true, b1, acc, false, false);
                #pragma unroll
                for (int j = 0; j < 8; ++j) {
                    int m = mtile * 16 + j + 8 * half;
                    int n = nt * 16 + l15;
                    lds_logits[m * VOCAB + n] = acc[j];
                }
            }
            __syncthreads();
            if (tid < BATCH) {
                const int b   = tid;
                const int tok = tokens[b * SEQ + t];
                const int* row = lds_logits + b * VOCAB;
                float mx = -1e30f;
                for (int n = 0; n < VOCAB; ++n) {
                    float v = (float)row[n] * LOGIT_SCALE;
                    mx = fmaxf(mx, v);
                }
                float s = 0.0f;
                for (int n = 0; n < VOCAB; ++n)
                    s += expf((float)row[n] * LOGIT_SCALE - mx);
                loss_acc += mx + logf(s) - (float)row[tok] * LOGIT_SCALE;
            }
            for (int i = tid; i < BATCH * READ_DIM; i += NTHREADS) {
                int b = i >> 7, j = i & 127;
                int tok = tokens[b * SEQ + t];
                X0[b * D_MODEL + CARRY + j] = embedS[tok * READ_DIM + j];
            }
        }
        grid_barrier(cnt, gen);
    }

    if (wg == 0) {
        if (tid < BATCH) lds_loss[tid] = loss_acc;
        __syncthreads();
        if (tid == 0) {
            float s = 0.0f;
            for (int i = 0; i < BATCH; ++i) s += lds_loss[i];
            out[0] = s / (float)(BATCH * SEQ);
        }
    }
}

extern "C" void kernel_launch(void* const* d_in, const int* in_sizes, int n_in,
                              void* d_out, int out_size, void* d_ws, size_t ws_size,
                              hipStream_t stream) {
    const int*   tokens  = (const int*)  d_in[0];
    const float* initial = (const float*)d_in[1];
    const float* embed   = (const float*)d_in[2];
    const float* ff      = (const float*)d_in[3];
    const float* head    = (const float*)d_in[4];
    const float* thresh  = (const float*)d_in[5];

    char* ws = (char*)d_ws;
    signed char* WtS    = (signed char*)(ws + OFF_WT);
    signed char* embedS = (signed char*)(ws + OFF_EMBED);
    signed char* headTS = (signed char*)(ws + OFF_HEADT);
    int*         thrI   = (int*)        (ws + OFF_THR);
    signed char* X0     = (signed char*)(ws + OFF_X0);
    signed char* X1     = (signed char*)(ws + OFF_X1);
    signed char* READ   = (signed char*)(ws + OFF_READ);
    unsigned*    bar    = (unsigned*)   (ws + OFF_BAR);

    init_barrier_k<<<1, 1, 0, stream>>>(bar);
    prep_ff_k<<<(6 * 1024 * 1024) / NTHREADS, NTHREADS, 0, stream>>>(ff, WtS);
    prep_small_k<<<408, NTHREADS, 0, stream>>>(initial, embed, head, thresh,
                                               embedS, headTS, thrI, X0);
    brnn_main_k<<<NWG, NTHREADS, 0, stream>>>(tokens, WtS, embedS, headTS, thrI,
                                              X0, X1, READ, bar, (float*)d_out);
}
